// QuadAttention_71116068487401
// MI455X (gfx1250) — compile-verified, matrix operand order checked by reading, not run on hardware
//
#include <hip/hip_runtime.h>
#include <hip/hip_bf16.h>
#include <math.h>

#define BB 2
#define NN 2048
#define CC 256
#define HH 16
#define DD 16
#define MTOT (BB * NN) // 4096

typedef __bf16 bf16;
typedef __attribute__((ext_vector_type(8)))  __bf16 bf16x8;
typedef __attribute__((ext_vector_type(16))) __bf16 bf16x16;
typedef __attribute__((ext_vector_type(8)))  float  f32x8;

struct Ptr7 { const float* p[7]; };
struct Ptr8 { const float* p[8]; };

// ---------------------------------------------------------------------------
// Fragment loader: matrix stored "lane-row-major" (row r contiguous in K).
// CDNA5 16-bit A/B layout: lanes 0-15 row=lane, K = kb+{0..7,16..23};
// lanes 16-31 row=lane-16, K = kb+{8..15,24..31}.
// ---------------------------------------------------------------------------
__device__ __forceinline__ bf16x16 load_frag(const bf16* __restrict__ base,
                                             int ld, int row, int kb, int lane) {
  const bf16* p = base + (size_t)(row + (lane & 15)) * ld + kb + ((lane >> 4) << 3);
  bf16x8 lo = *reinterpret_cast<const bf16x8*>(p);
  bf16x8 hi = *reinterpret_cast<const bf16x8*>(p + 16);
  bf16x16 f;
#pragma unroll
  for (int e = 0; e < 8; ++e) { f[e] = lo[e]; f[e + 8] = hi[e]; }
  return f;
}

__device__ __forceinline__ f32x8 wmma_bf16(bf16x16 a, bf16x16 b, f32x8 c) {
  return __builtin_amdgcn_wmma_f32_16x16x32_bf16(false, a, false, b, (short)0, c,
                                                 false, false);
}

// 32x32 register-blocked GEMM core: D[2][2] 16x16 tiles, A,B lane-row-major.
// 4 WMMAs per 4 fragment loads per k-step, prefetch of next k-block.
__device__ __forceinline__ void gemm_tile32(const bf16* __restrict__ A, int lda,
                                            const bf16* __restrict__ B, int ldb,
                                            int m0, int n0, int K, int lane,
                                            f32x8 acc[2][2]) {
#pragma unroll 2
  for (int kb = 0; kb < K; kb += 32) {
    bf16x16 a0 = load_frag(A, lda, m0, kb, lane);
    bf16x16 a1 = load_frag(A, lda, m0 + 16, kb, lane);
    bf16x16 b0 = load_frag(B, ldb, n0, kb, lane);
    bf16x16 b1 = load_frag(B, ldb, n0 + 16, kb, lane);
    if (kb + 32 < K) { // gfx1250 global_prefetch_b8 of next k-block
      __builtin_prefetch(A + (size_t)(m0 + (lane & 15)) * lda + kb + 32, 0, 3);
      __builtin_prefetch(B + (size_t)(n0 + (lane & 15)) * ldb + kb + 32, 0, 3);
    }
    acc[0][0] = wmma_bf16(a0, b0, acc[0][0]);
    acc[0][1] = wmma_bf16(a0, b1, acc[0][1]);
    acc[1][0] = wmma_bf16(a1, b0, acc[1][0]);
    acc[1][1] = wmma_bf16(a1, b1, acc[1][1]);
  }
}

__device__ __forceinline__ void zero_acc(f32x8 acc[2][2]) {
#pragma unroll
  for (int tm = 0; tm < 2; ++tm)
#pragma unroll
    for (int tn = 0; tn < 2; ++tn)
#pragma unroll
      for (int r = 0; r < 8; ++r) acc[tm][tn][r] = 0.f;
}

// ---------------------------------------------------------------------------
// Conversions
// ---------------------------------------------------------------------------
__global__ void convx_kernel(const float* __restrict__ x, bf16* __restrict__ xh) {
  int t = blockIdx.x * blockDim.x + threadIdx.x;
  xh[t] = (bf16)x[t];
}

__global__ void convw_kernel(Ptr8 w, bf16* __restrict__ wh) {
  int t = blockIdx.x * blockDim.x + threadIdx.x; // 0..CC*CC
  int z = blockIdx.z;
  wh[(size_t)z * CC * CC + t] = (bf16)w.p[z][t];
}

// ---------------------------------------------------------------------------
// 7 fused residual linears: lin[z] = (x + x@W_z^T + b_z) * mask * scale_z
// One wave per 32x32 tile. B[k,n] = W[n,k] -> row-major W rows are K-contiguous.
// ---------------------------------------------------------------------------
__global__ void lin7_kernel(const bf16* __restrict__ xh, const bf16* __restrict__ wh,
                            const float* __restrict__ x, const float* __restrict__ mask,
                            Ptr7 bias, float scale, float* __restrict__ lin) {
  const int lane = threadIdx.x;
  const int mt = blockIdx.x, nt = blockIdx.y, z = blockIdx.z;
  const bf16* W = wh + (size_t)z * CC * CC;
  f32x8 acc[2][2];
  zero_acc(acc);
  gemm_tile32(xh, CC, W, CC, mt * 32, nt * 32, CC, lane, acc);

  const float* bz = bias.p[z];
  const float sc = (z >= 3) ? scale : 1.0f;
  float* outz = lin + (size_t)z * MTOT * CC;
#pragma unroll
  for (int tm = 0; tm < 2; ++tm)
#pragma unroll
    for (int tn = 0; tn < 2; ++tn) {
      const int col = nt * 32 + tn * 16 + (lane & 15);
#pragma unroll
      for (int r = 0; r < 8; ++r) {
        int row = mt * 32 + tm * 16 + r + ((lane >> 4) << 3);
        float v = acc[tm][tn][r] + x[(size_t)row * CC + col] + bz[col];
        outz[(size_t)row * CC + col] = v * mask[row] * sc;
      }
    }
}

// ---------------------------------------------------------------------------
// Outer products for one (b,h): A1t[p,n]=k1_i*v_j, A2t[p,n]=k2_k*k3_l (256xN),
// Qt[n,p]=q2_k*q3_l (Nx256), p = first*16 + second.
// ---------------------------------------------------------------------------
__global__ void outer_kernel(const float* __restrict__ lin, int b, int h,
                             bf16* __restrict__ A1t, bf16* __restrict__ A2t,
                             bf16* __restrict__ Qt) {
  int t = blockIdx.x * blockDim.x + threadIdx.x; // NN*256
  int n = t >> 8, p = t & 255;
  int i = p >> 4, j = p & 15;
  size_t base = ((size_t)(b * NN + n)) * CC + h * DD;
  const float* q2 = lin + (size_t)1 * MTOT * CC;
  const float* q3 = lin + (size_t)2 * MTOT * CC;
  const float* k1 = lin + (size_t)3 * MTOT * CC;
  const float* k2 = lin + (size_t)4 * MTOT * CC;
  const float* k3 = lin + (size_t)5 * MTOT * CC;
  const float* vv = lin + (size_t)6 * MTOT * CC;
  A1t[(size_t)p * NN + n] = (bf16)(k1[base + i] * vv[base + j]);
  A2t[(size_t)p * NN + n] = (bf16)(k2[base + i] * k3[base + j]);
  Qt[(size_t)n * CC + p]  = (bf16)(q2[base + i] * q3[base + j]);
}

// ---------------------------------------------------------------------------
// state[p,q] = sum_n A1t[p,n] * A2t[q,n]  (256x256, K = 2048), stored bf16
// ---------------------------------------------------------------------------
__global__ void state_kernel(const bf16* __restrict__ A1t, const bf16* __restrict__ A2t,
                             bf16* __restrict__ stateh) {
  const int lane = threadIdx.x;
  const int mt = blockIdx.x, nt = blockIdx.y;
  f32x8 acc[2][2];
  zero_acc(acc);
  gemm_tile32(A1t, NN, A2t, NN, mt * 32, nt * 32, NN, lane, acc);
#pragma unroll
  for (int tm = 0; tm < 2; ++tm)
#pragma unroll
    for (int tn = 0; tn < 2; ++tn) {
      const int col = nt * 32 + tn * 16 + (lane & 15);
#pragma unroll
      for (int r = 0; r < 8; ++r) {
        int row = mt * 32 + tm * 16 + r + ((lane >> 4) << 3);
        stateh[(size_t)row * CC + col] = (bf16)acc[tm][tn][r];
      }
    }
}

// ---------------------------------------------------------------------------
// T[n,ij] = sum_kl Qt[n,kl] * state[ij,kl]  (2048x256, K = 256)
// B[k,col] = state[col,k] -> state rows are K-contiguous.
// ---------------------------------------------------------------------------
__global__ void tgemm_kernel(const bf16* __restrict__ Qt, const bf16* __restrict__ stateh,
                             float* __restrict__ T) {
  const int lane = threadIdx.x;
  const int mt = blockIdx.x, nt = blockIdx.y;
  f32x8 acc[2][2];
  zero_acc(acc);
  gemm_tile32(Qt, CC, stateh, CC, mt * 32, nt * 32, CC, lane, acc);
#pragma unroll
  for (int tm = 0; tm < 2; ++tm)
#pragma unroll
    for (int tn = 0; tn < 2; ++tn) {
      const int col = nt * 32 + tn * 16 + (lane & 15);
#pragma unroll
      for (int r = 0; r < 8; ++r) {
        int row = mt * 32 + tm * 16 + r + ((lane >> 4) << 3);
        T[(size_t)row * CC + col] = acc[tm][tn][r];
      }
    }
}

// ---------------------------------------------------------------------------
// attn[b,n,h*16+j] = sum_i q1[b,n,h*16+i] * T[n, i*16+j]
// ---------------------------------------------------------------------------
__global__ void combine_kernel(const float* __restrict__ lin, const float* __restrict__ T,
                               int b, int h, float* __restrict__ attn) {
  int t = blockIdx.x * blockDim.x + threadIdx.x; // NN*16
  int n = t >> 4, j = t & 15;
  size_t base = ((size_t)(b * NN + n)) * CC + h * DD;
  const float* q1 = lin; // z = 0
  float s = 0.f;
#pragma unroll
  for (int i = 0; i < 16; ++i) s += q1[base + i] * T[(size_t)n * CC + i * 16 + j];
  attn[base + j] = s;
}

// ---------------------------------------------------------------------------
// LayerNorm over C=256 per token; one block per row
// ---------------------------------------------------------------------------
__global__ void ln_kernel(const float* __restrict__ attn, const float* __restrict__ gamma,
                          const float* __restrict__ beta, bf16* __restrict__ yln) {
  __shared__ float r1[256];
  __shared__ float r2[256];
  const int row = blockIdx.x, c = threadIdx.x;
  float v = attn[(size_t)row * CC + c];
  r1[c] = v;
  r2[c] = v * v;
  __syncthreads();
  for (int s = 128; s > 0; s >>= 1) {
    if (c < s) { r1[c] += r1[c + s]; r2[c] += r2[c + s]; }
    __syncthreads();
  }
  float mu = r1[0] * (1.0f / CC);
  float var = r2[0] * (1.0f / CC) - mu * mu;
  float y = (v - mu) * rsqrtf(var + 1e-5f) * gamma[c] + beta[c];
  yln[(size_t)row * CC + c] = (bf16)y;
}

// ---------------------------------------------------------------------------
// Final linear: out = yln @ Wo^T + bo (fp32 output)
// ---------------------------------------------------------------------------
__global__ void final_kernel(const bf16* __restrict__ yln, const bf16* __restrict__ Wo,
                             const float* __restrict__ bo, float* __restrict__ out) {
  const int lane = threadIdx.x;
  const int mt = blockIdx.x, nt = blockIdx.y;
  f32x8 acc[2][2];
  zero_acc(acc);
  gemm_tile32(yln, CC, Wo, CC, mt * 32, nt * 32, CC, lane, acc);
#pragma unroll
  for (int tm = 0; tm < 2; ++tm)
#pragma unroll
    for (int tn = 0; tn < 2; ++tn) {
      const int col = nt * 32 + tn * 16 + (lane & 15);
#pragma unroll
      for (int r = 0; r < 8; ++r) {
        int row = mt * 32 + tm * 16 + r + ((lane >> 4) << 3);
        out[(size_t)row * CC + col] = acc[tm][tn][r] + bo[col];
      }
    }
}

// ---------------------------------------------------------------------------
extern "C" void kernel_launch(void* const* d_in, const int* in_sizes, int n_in,
                              void* d_out, int out_size, void* d_ws, size_t ws_size,
                              hipStream_t stream) {
  (void)in_sizes; (void)n_in; (void)out_size; (void)ws_size;

  const float* x    = (const float*)d_in[0];
  const float* mask = (const float*)d_in[1];
  Ptr8 w;
  w.p[0] = (const float*)d_in[2];   // Wq1
  w.p[1] = (const float*)d_in[4];   // Wq2
  w.p[2] = (const float*)d_in[6];   // Wq3
  w.p[3] = (const float*)d_in[8];   // Wk1
  w.p[4] = (const float*)d_in[10];  // Wk2
  w.p[5] = (const float*)d_in[12];  // Wk3
  w.p[6] = (const float*)d_in[14];  // Wv
  w.p[7] = (const float*)d_in[16];  // Wo
  Ptr7 bias;
  bias.p[0] = (const float*)d_in[3];
  bias.p[1] = (const float*)d_in[5];
  bias.p[2] = (const float*)d_in[7];
  bias.p[3] = (const float*)d_in[9];
  bias.p[4] = (const float*)d_in[11];
  bias.p[5] = (const float*)d_in[13];
  bias.p[6] = (const float*)d_in[15];
  const float* bo    = (const float*)d_in[17];
  const float* gamma = (const float*)d_in[18];
  const float* beta  = (const float*)d_in[19];
  float* out = (float*)d_out;

  // Workspace carve-up (~44 MB total), 256B aligned
  char* ws = (char*)d_ws;
  size_t off = 0;
  auto alloc = [&](size_t bytes) -> void* {
    void* p = ws + off;
    off = (off + bytes + 255) & ~(size_t)255;
    return p;
  };
  bf16*  xh     = (bf16*)alloc((size_t)MTOT * CC * 2);
  bf16*  wh     = (bf16*)alloc((size_t)8 * CC * CC * 2);
  float* lin    = (float*)alloc((size_t)7 * MTOT * CC * 4);
  bf16*  A1t    = (bf16*)alloc((size_t)CC * NN * 2);
  bf16*  A2t    = (bf16*)alloc((size_t)CC * NN * 2);
  bf16*  Qt     = (bf16*)alloc((size_t)NN * CC * 2);
  bf16*  stateh = (bf16*)alloc((size_t)CC * CC * 2);
  float* T      = (float*)alloc((size_t)NN * CC * 4);
  float* attn   = (float*)alloc((size_t)MTOT * CC * 4);
  bf16*  yln    = (bf16*)alloc((size_t)MTOT * CC * 2);

  const float scale = powf((float)NN, -0.25f);

  convx_kernel<<<dim3(MTOT * CC / 256), 256, 0, stream>>>(x, xh);
  convw_kernel<<<dim3(CC * CC / 256, 1, 8), 256, 0, stream>>>(w, wh);

  lin7_kernel<<<dim3(MTOT / 32, CC / 32, 7), 32, 0, stream>>>(xh, wh, x, mask, bias,
                                                              scale, lin);

  for (int b = 0; b < BB; ++b) {
    for (int h = 0; h < HH; ++h) {
      outer_kernel<<<dim3(NN * CC / 256), 256, 0, stream>>>(lin, b, h, A1t, A2t, Qt);
      state_kernel<<<dim3(CC / 32, CC / 32), 32, 0, stream>>>(A1t, A2t, stateh);
      tgemm_kernel<<<dim3(NN / 32, CC / 32), 32, 0, stream>>>(Qt, stateh, T);
      combine_kernel<<<dim3(NN * 16 / 256), 256, 0, stream>>>(lin, T, b, h, attn);
    }
  }

  ln_kernel<<<dim3(MTOT), 256, 0, stream>>>(attn, gamma, beta, yln);
  final_kernel<<<dim3(MTOT / 32, CC / 32), 32, 0, stream>>>(yln, wh + (size_t)7 * CC * CC,
                                                            bo, out);
}
